// ChemicallyInformedRhoFold_67619965108970
// MI455X (gfx1250) — matrix operand order repaired and checked by reading, not compiled, not run on hardware
//
#include <hip/hip_runtime.h>
#include <hip/hip_bf16.h>

typedef __attribute__((ext_vector_type(16))) __bf16 v16bf;
typedef __attribute__((ext_vector_type(8)))  float  v8f;

#define WMMA_BF16(a, b, c) \
    __builtin_amdgcn_wmma_f32_16x16x32_bf16(false, (a), false, (b), (short)0, (c), false, false)

// ---------------------------------------------------------------- helpers ---

__device__ __forceinline__ unsigned short f32_to_bf16_bits(float f) {
    unsigned int u = __float_as_uint(f);
    unsigned int r = u + 0x7FFFu + ((u >> 16) & 1u);   // round-to-nearest-even
    return (unsigned short)(r >> 16);
}
__device__ __forceinline__ float bf16_lo(unsigned int w) {
    return __uint_as_float(w << 16);
}
__device__ __forceinline__ float bf16_hi(unsigned int w) {
    return __uint_as_float(w & 0xFFFF0000u);
}
__device__ __forceinline__ unsigned int pack2bf(float a, float b) {
    return (unsigned int)f32_to_bf16_bits(a) | ((unsigned int)f32_to_bf16_bits(b) << 16);
}

// Packed-A layout (CDNA5 16-bit A 16x32 fragment, ISA 7.12.2):
// element (m16, kk32) of a 16x32 tile lives at [lane][idx].
// NOTE: 8 consecutive k (8-aligned) are contiguous in idx.
__device__ __forceinline__ void a_tile_pos(int m16, int kk, int& lane, int& idx) {
    lane = m16 + (((kk >> 3) & 1) << 4);
    idx  = ((kk >> 4) << 3) + (((kk >> 1) & 3) << 1) + (kk & 1);
}
__device__ __forceinline__ size_t a_pack_off(int m, int k, int K) {
    int lane, idx;
    a_tile_pos(m & 15, k & 31, lane, idx);
    return ((size_t)((m >> 4) * (K >> 5) + (k >> 5)) * 32 + lane) * 16 + idx;
}
// Packed-B layout (32x16 fragment): element (kk32, n16) at [lane][idx].
__device__ __forceinline__ size_t b_pack_off(int k, int n, int K) {
    int lane = (n & 15) + (((k >> 3) & 1) << 4);
    int idx  = (((k & 31) >> 4) << 3) + (k & 7);
    return ((size_t)((n >> 4) * (K >> 5) + (k >> 5)) * 32 + lane) * 16 + idx;
}

// ---------------------------------------------------------------- kernels ---

__global__ void pack_a_kernel(const float* __restrict__ src,
                              unsigned short* __restrict__ dst, int M, int K) {
    int i = blockIdx.x * blockDim.x + threadIdx.x;
    if (i >= M * K) return;
    int m = i / K, k = i - m * K;
    dst[a_pack_off(m, k, K)] = f32_to_bf16_bits(src[i]);
}

__global__ void pack_b_kernel(const float* __restrict__ src,
                              unsigned short* __restrict__ dst, int K, int N) {
    int i = blockIdx.x * blockDim.x + threadIdx.x;
    if (i >= K * N) return;
    int k = i / N, n = i - k * N;
    dst[b_pack_off(k, n, K)] = f32_to_bf16_bits(src[i]);
}

// GEMM: C(MxN) = A@B [+bias][relu], A/B pre-packed bf16. Wave = 16x32 C tile.
__global__ __launch_bounds__(256)
void gemm_bf16_kernel(const unsigned short* __restrict__ Apk,
                      const unsigned short* __restrict__ Bpk,
                      const float* __restrict__ bias,
                      float* __restrict__ Cf,
                      int M, int N, int K, int relu) {
    int lane = threadIdx.x & 31;
    int wave = threadIdx.x >> 5;
    int KT   = K >> 5;
    int nPair = N >> 5;
    int tile  = blockIdx.x * 8 + wave;
    if (tile >= (M >> 4) * nPair) return;      // wave-uniform
    int tm = tile / nPair, tp = tile - tm * nPair;
    int tn0 = tp * 2, tn1 = tn0 + 1;

    const v16bf* A  = reinterpret_cast<const v16bf*>(Apk) + (size_t)tm  * KT * 32 + lane;
    const v16bf* B0 = reinterpret_cast<const v16bf*>(Bpk) + (size_t)tn0 * KT * 32 + lane;
    const v16bf* B1 = reinterpret_cast<const v16bf*>(Bpk) + (size_t)tn1 * KT * 32 + lane;

    v8f acc0 = {0.f,0.f,0.f,0.f,0.f,0.f,0.f,0.f};
    v8f acc1 = {0.f,0.f,0.f,0.f,0.f,0.f,0.f,0.f};
    for (int kt = 0; kt < KT; ++kt) {
        v16bf a  = A[(size_t)kt * 32];
        v16bf b0 = B0[(size_t)kt * 32];
        v16bf b1 = B1[(size_t)kt * 32];
        acc0 = WMMA_BF16(a, b0, acc0);
        acc1 = WMMA_BF16(a, b1, acc1);
    }
    int n0 = tn0 * 16 + (lane & 15);
    int n1 = n0 + 16;
    float bv0 = bias ? bias[n0] : 0.f;
    float bv1 = bias ? bias[n1] : 0.f;
#pragma unroll
    for (int r = 0; r < 8; ++r) {
        int m = tm * 16 + r + ((lane >> 4) << 3);
        float v0 = acc0[r] + bv0, v1 = acc1[r] + bv1;
        if (relu) { v0 = fmaxf(v0, 0.f); v1 = fmaxf(v1, 0.f); }
        Cf[(size_t)m * N + n0] = v0;
        Cf[(size_t)m * N + n1] = v1;
    }
}

// Row LayerNorm; optional f32 out (row-major) and/or bf16 out (packed-A).
__global__ __launch_bounds__(256)
void ln_rows_kernel(const float* __restrict__ X, const float* __restrict__ g,
                    const float* __restrict__ be, float* __restrict__ Yf,
                    unsigned short* __restrict__ Ypk, int N) {
    int row = blockIdx.x;
    const float* x = X + (size_t)row * N;
    __shared__ float rs[256], rq[256];
    float s = 0.f, q = 0.f;
    for (int k = threadIdx.x; k < N; k += 256) { float v = x[k]; s += v; q += v * v; }
    rs[threadIdx.x] = s; rq[threadIdx.x] = q;
    __syncthreads();
    for (int off = 128; off > 0; off >>= 1) {
        if (threadIdx.x < off) {
            rs[threadIdx.x] += rs[threadIdx.x + off];
            rq[threadIdx.x] += rq[threadIdx.x + off];
        }
        __syncthreads();
    }
    float mean = rs[0] / N;
    float var  = rq[0] / N - mean * mean;
    float rstd = rsqrtf(var + 1e-5f);
    for (int k = threadIdx.x; k < N; k += 256) {
        float y = (x[k] - mean) * rstd * g[k] + be[k];
        if (Yf)  Yf[(size_t)row * N + k] = y;
        if (Ypk) Ypk[a_pack_off(row, k, N)] = f32_to_bf16_bits(y);
    }
}

// dms & shape scalar heads.
__global__ __launch_bounds__(160)
void heads_kernel(const float* __restrict__ sh,
                  const float* __restrict__ w1d, const float* __restrict__ b1d,
                  const float* __restrict__ w2d, const float* __restrict__ b2d,
                  const float* __restrict__ w1s, const float* __restrict__ b1s,
                  const float* __restrict__ w2s, const float* __restrict__ b2s,
                  float* __restrict__ dms_out, float* __restrict__ shp_out) {
    int row = blockIdx.x;
    const float* x = sh + (size_t)row * 320;
    __shared__ float hd[160], hs[160];
    int t = threadIdx.x;
    float a = b1d[t], b = b1s[t];
    for (int k = 0; k < 320; ++k) {
        float xv = x[k];
        a += xv * w1d[k * 160 + t];
        b += xv * w1s[k * 160 + t];
    }
    hd[t] = fmaxf(a, 0.f) * w2d[t];
    hs[t] = fmaxf(b, 0.f) * w2s[t];
    __syncthreads();
    if (t == 0) {
        float sd = b2d[0], ss = b2s[0];
        for (int k = 0; k < 160; ++k) { sd += hd[k]; ss += hs[k]; }
        dms_out[row] = sd;
        shp_out[row] = ss;
    }
}

// Pair head. Block = (i, 64-j tile), 256 threads (8 waves).
// Phase A: h = relu(hi[i]+hj[j]+chem@Wc+b1) -> packed bf16 LDS (vectorized),
//          stats in f32, in-place LN rescale (b128 LDS passes).
// Phase B: (64x512)@(512x256) relu -> LDS  (wave = 4 M-tiles x 2 N-tiles).
// Phase C: (64x256)@(256x128) -> pair      (wave = 4 M-tiles x 1 N-tile).
__global__ __launch_bounds__(256)
void pair_kernel(const float* __restrict__ hi, const float* __restrict__ hj,
                 const float* __restrict__ dms, const float* __restrict__ shp,
                 const float* __restrict__ Wc,            // 4 x 512
                 const float* __restrict__ b1, const float* __restrict__ g,
                 const float* __restrict__ be,
                 const unsigned short* __restrict__ w2pk, // packed 512x256
                 const float* __restrict__ b2,
                 const unsigned short* __restrict__ w3pk, // packed 256x128
                 const float* __restrict__ b3,
                 float* __restrict__ pair_out) {
    __shared__ __attribute__((aligned(32))) unsigned short hbfp[4][16][32][16]; // 64KB
    __shared__ __attribute__((aligned(32))) unsigned short gbfp[4][8][32][16];  // 32KB
    __shared__ float redS[64][4], redQ[64][4];

    int i   = blockIdx.x;
    int jt  = blockIdx.y;
    int tid = threadIdx.x;
    int m   = tid >> 2;     // 0..63 row within j-tile
    int t4  = tid & 3;
    int j   = jt * 64 + m;
    int mt  = m >> 4, m16 = m & 15;

    // Warm the cache with this block's W2 stream (global_prefetch_b8).
    {
        const char* p = reinterpret_cast<const char*>(w2pk);
        for (int off = tid; off < 2048; off += 256)
            __builtin_prefetch(p + (size_t)off * 128, 0, 1);
    }

    float di = dms[i], dj = dms[j], si = shp[i], sj = shp[j];
    float c0 = di - dj, c1 = di * dj, c2 = si - sj, c3 = si * sj;

    const float4* HI = reinterpret_cast<const float4*>(hi + (size_t)i * 512);
    const float4* HJ = reinterpret_cast<const float4*>(hj + (size_t)j * 512);
    const float4* WC0 = reinterpret_cast<const float4*>(Wc);
    const float4* WC1 = reinterpret_cast<const float4*>(Wc + 512);
    const float4* WC2 = reinterpret_cast<const float4*>(Wc + 1024);
    const float4* WC3 = reinterpret_cast<const float4*>(Wc + 1536);
    const float4* B1v = reinterpret_cast<const float4*>(b1);

    // --- Phase A: relu(h) -> packed bf16 (8 k per b128 store), f32 stats ----
    float s = 0.f, q = 0.f;
    for (int kb = t4 * 128; kb < t4 * 128 + 128; kb += 8) {
        unsigned int pk[4];
#pragma unroll
        for (int h = 0; h < 2; ++h) {
            int q4 = (kb >> 2) + h;
            float4 fa = HI[q4], fb = HJ[q4];
            float4 w0 = WC0[q4], w1 = WC1[q4], w2 = WC2[q4], w3 = WC3[q4];
            float4 bb = B1v[q4];
            float vx = fmaxf(fa.x + fb.x + c0*w0.x + c1*w1.x + c2*w2.x + c3*w3.x + bb.x, 0.f);
            float vy = fmaxf(fa.y + fb.y + c0*w0.y + c1*w1.y + c2*w2.y + c3*w3.y + bb.y, 0.f);
            float vz = fmaxf(fa.z + fb.z + c0*w0.z + c1*w1.z + c2*w2.z + c3*w3.z + bb.z, 0.f);
            float vw = fmaxf(fa.w + fb.w + c0*w0.w + c1*w1.w + c2*w2.w + c3*w3.w + bb.w, 0.f);
            s += vx + vy + vz + vw;
            q += vx*vx + vy*vy + vz*vz + vw*vw;
            pk[h*2+0] = pack2bf(vx, vy);
            pk[h*2+1] = pack2bf(vz, vw);
        }
        int kk   = kb & 31;
        int lane = m16 + (((kk >> 3) & 1) << 4);
        int idx8 = (kk >> 4) << 3;
        uint4* dst = reinterpret_cast<uint4*>(&hbfp[mt][kb >> 5][lane][idx8]);
        uint4 P; P.x = pk[0]; P.y = pk[1]; P.z = pk[2]; P.w = pk[3];
        *dst = P;
    }
    redS[m][t4] = s; redQ[m][t4] = q;
    __syncthreads();
    float S = redS[m][0] + redS[m][1] + redS[m][2] + redS[m][3];
    float Q = redQ[m][0] + redQ[m][1] + redQ[m][2] + redQ[m][3];
    float mean = S * (1.f / 512.f);
    float var  = Q * (1.f / 512.f) - mean * mean;
    float rstd = rsqrtf(var + 1e-5f);

    // in-place LN rescale (b128 LDS load/modify/store)
    const float4* Gv  = reinterpret_cast<const float4*>(g);
    const float4* BEv = reinterpret_cast<const float4*>(be);
    for (int kb = t4 * 128; kb < t4 * 128 + 128; kb += 8) {
        int kk   = kb & 31;
        int lane = m16 + (((kk >> 3) & 1) << 4);
        int idx8 = (kk >> 4) << 3;
        uint4* ptr = reinterpret_cast<uint4*>(&hbfp[mt][kb >> 5][lane][idx8]);
        uint4 P = *ptr;
        unsigned int w[4] = {P.x, P.y, P.z, P.w};
#pragma unroll
        for (int h = 0; h < 2; ++h) {
            int q4 = (kb >> 2) + h;
            float4 gg = Gv[q4], ee = BEv[q4];
            float x0 = bf16_lo(w[h*2+0]), x1 = bf16_hi(w[h*2+0]);
            float x2 = bf16_lo(w[h*2+1]), x3 = bf16_hi(w[h*2+1]);
            w[h*2+0] = pack2bf((x0 - mean) * rstd * gg.x + ee.x,
                               (x1 - mean) * rstd * gg.y + ee.y);
            w[h*2+1] = pack2bf((x2 - mean) * rstd * gg.z + ee.z,
                               (x3 - mean) * rstd * gg.w + ee.w);
        }
        P.x = w[0]; P.y = w[1]; P.z = w[2]; P.w = w[3];
        *ptr = P;
    }
    __syncthreads();

    int wave = tid >> 5, lane = tid & 31;
    const v16bf* HB = reinterpret_cast<const v16bf*>(&hbfp[0][0][0][0]);
    const v16bf* GB = reinterpret_cast<const v16bf*>(&gbfp[0][0][0][0]);
    const v16bf* W2 = reinterpret_cast<const v16bf*>(w2pk);
    const v16bf* W3 = reinterpret_cast<const v16bf*>(w3pk);

    // --- Phase B: (64x512)@(512x256), relu -> gbfp (4x2 wave tiling) --------
    {
        int nt0 = wave * 2, nt1 = nt0 + 1;
        v8f acc[4][2];
#pragma unroll
        for (int a = 0; a < 4; ++a)
#pragma unroll
            for (int b = 0; b < 2; ++b)
                acc[a][b] = (v8f){0.f,0.f,0.f,0.f,0.f,0.f,0.f,0.f};
        for (int kt = 0; kt < 16; ++kt) {
            v16bf fb0 = W2[((size_t)nt0 * 16 + kt) * 32 + lane];
            v16bf fb1 = W2[((size_t)nt1 * 16 + kt) * 32 + lane];
#pragma unroll
            for (int a = 0; a < 4; ++a) {
                v16bf fa = HB[(size_t)(a * 16 + kt) * 32 + lane];
                acc[a][0] = WMMA_BF16(fa, fb0, acc[a][0]);
                acc[a][1] = WMMA_BF16(fa, fb1, acc[a][1]);
            }
        }
        int n0 = nt0 * 16 + (lane & 15);
        int n1 = n0 + 16;
        float bv0 = b2[n0], bv1 = b2[n1];
#pragma unroll
        for (int a = 0; a < 4; ++a) {
#pragma unroll
            for (int r = 0; r < 8; ++r) {
                int mr = r + ((lane >> 4) << 3);
                int la, ix;
                a_tile_pos(mr, n0 & 31, la, ix);
                gbfp[a][n0 >> 5][la][ix] = f32_to_bf16_bits(fmaxf(acc[a][0][r] + bv0, 0.f));
                a_tile_pos(mr, n1 & 31, la, ix);
                gbfp[a][n1 >> 5][la][ix] = f32_to_bf16_bits(fmaxf(acc[a][1][r] + bv1, 0.f));
            }
        }
    }
    __syncthreads();

    // --- Phase C: (64x256)@(256x128) -> pair (4x1 wave tiling) --------------
    {
        int nt = wave;  // 0..7
        v8f acc[4];
#pragma unroll
        for (int a = 0; a < 4; ++a)
            acc[a] = (v8f){0.f,0.f,0.f,0.f,0.f,0.f,0.f,0.f};
        for (int kt = 0; kt < 8; ++kt) {
            v16bf fb = W3[((size_t)nt * 8 + kt) * 32 + lane];
#pragma unroll
            for (int a = 0; a < 4; ++a) {
                v16bf fa = GB[(size_t)(a * 8 + kt) * 32 + lane];
                acc[a] = WMMA_BF16(fa, fb, acc[a]);
            }
        }
        int n = nt * 16 + (lane & 15);
        float bv = b3[n];
#pragma unroll
        for (int a = 0; a < 4; ++a) {
#pragma unroll
            for (int r = 0; r < 8; ++r) {
                int mr = r + ((lane >> 4) << 3);
                int jj = jt * 64 + a * 16 + mr;
                pair_out[((size_t)i * 384 + jj) * 128 + n] = acc[a][r] + bv;
            }
        }
    }
}

// ---------------------------------------------------------------- launch ----

extern "C" void kernel_launch(void* const* d_in, const int* in_sizes, int n_in,
                              void* d_out, int out_size, void* d_ws, size_t ws_size,
                              hipStream_t stream) {
    const float* features = (const float*)d_in[0];   // 384x640
    const float* sp_w  = (const float*)d_in[1];      // 640x320
    const float* sp_b  = (const float*)d_in[2];
    const float* sp_g  = (const float*)d_in[3];
    const float* sp_be = (const float*)d_in[4];
    const float* dms_w1 = (const float*)d_in[5];
    const float* dms_b1 = (const float*)d_in[6];
    const float* dms_w2 = (const float*)d_in[7];
    const float* dms_b2 = (const float*)d_in[8];
    const float* sh_w1 = (const float*)d_in[9];
    const float* sh_b1 = (const float*)d_in[10];
    const float* sh_w2 = (const float*)d_in[11];
    const float* sh_b2 = (const float*)d_in[12];
    const float* ts_w1 = (const float*)d_in[13];     // 640x512
    const float* ts_b1 = (const float*)d_in[14];
    const float* ts_g  = (const float*)d_in[15];
    const float* ts_be = (const float*)d_in[16];
    const float* ts_w2 = (const float*)d_in[17];     // 512x384
    const float* ts_b2 = (const float*)d_in[18];
    const float* tp_w1 = (const float*)d_in[19];     // 1284x512
    const float* tp_b1 = (const float*)d_in[20];
    const float* tp_g  = (const float*)d_in[21];
    const float* tp_be = (const float*)d_in[22];
    const float* tp_w2 = (const float*)d_in[23];     // 512x256
    const float* tp_b2 = (const float*)d_in[24];
    const float* tp_w3 = (const float*)d_in[25];     // 256x128
    const float* tp_b3 = (const float*)d_in[26];

    char* ws = (char*)d_ws;
    unsigned short* fpk    = (unsigned short*)(ws + 0);        // 384x640 A-packed
    unsigned short* spwp   = (unsigned short*)(ws + 491520);   // 640x320 B-packed
    unsigned short* tsw1p  = (unsigned short*)(ws + 901120);   // 640x512 B-packed
    unsigned short* tsw2p  = (unsigned short*)(ws + 1556480);  // 512x384 B-packed
    unsigned short* Wap    = (unsigned short*)(ws + 1949696);  // 640x512 B-packed
    unsigned short* Wbp    = (unsigned short*)(ws + 2605056);  // 640x512 B-packed
    unsigned short* w2p    = (unsigned short*)(ws + 3260416);  // 512x256 B-packed
    unsigned short* w3p    = (unsigned short*)(ws + 3522560);  // 256x128 B-packed
    float* s0      = (float*)(ws + 3588096);                   // 384x320
    float* sharedF = (float*)(ws + 4079616);                   // 384x320
    float* t0      = (float*)(ws + 4571136);                   // 384x512
    unsigned short* tpk = (unsigned short*)(ws + 5357568);     // 384x512 A-packed
    float* hiF     = (float*)(ws + 5750784);                   // 384x512
    float* hjF     = (float*)(ws + 6537216);                   // 384x512

    float* out_single = (float*)d_out;                         // 384x384
    float* out_pair   = out_single + 147456;                   // 384x384x128
    float* out_dms    = out_single + 19021824;                 // 384
    float* out_shape  = out_single + 19022208;                 // 384

    auto packA = [&](const float* src, unsigned short* dst, int M, int K) {
        pack_a_kernel<<<(M * K + 255) / 256, 256, 0, stream>>>(src, dst, M, K);
    };
    auto packB = [&](const float* src, unsigned short* dst, int K, int N) {
        pack_b_kernel<<<(K * N + 255) / 256, 256, 0, stream>>>(src, dst, K, N);
    };
    packA(features, fpk, 384, 640);
    packB(sp_w,  spwp,  640, 320);
    packB(ts_w1, tsw1p, 640, 512);
    packB(ts_w2, tsw2p, 512, 384);
    packB(tp_w1,             Wap, 640, 512);
    packB(tp_w1 + 640 * 512, Wbp, 640, 512);
    packB(tp_w2, w2p, 512, 256);
    packB(tp_w3, w3p, 256, 128);

    auto gemm = [&](const unsigned short* A, const unsigned short* B,
                    const float* bias, float* Cf, int M, int N, int K, int relu) {
        int tiles = (M / 16) * (N / 32);
        gemm_bf16_kernel<<<(tiles + 7) / 8, 256, 0, stream>>>(A, B, bias, Cf,
                                                              M, N, K, relu);
    };

    // shared = LN(relu(features @ sp_w + sp_b))
    gemm(fpk, spwp, sp_b, s0, 384, 320, 640, 1);
    ln_rows_kernel<<<384, 256, 0, stream>>>(s0, sp_g, sp_be, sharedF, nullptr, 320);

    // dms / shape heads
    heads_kernel<<<384, 160, 0, stream>>>(sharedF, dms_w1, dms_b1, dms_w2, dms_b2,
                                          sh_w1, sh_b1, sh_w2, sh_b2,
                                          out_dms, out_shape);

    // single = LN(relu(features @ ts_w1 + ts_b1)) @ ts_w2 + ts_b2
    gemm(fpk, tsw1p, ts_b1, t0, 384, 512, 640, 1);
    ln_rows_kernel<<<384, 256, 0, stream>>>(t0, ts_g, ts_be, nullptr, tpk, 512);
    gemm(tpk, tsw2p, ts_b2, out_single, 384, 384, 512, 0);

    // hi = features @ Wa ; hj = features @ Wb
    gemm(fpk, Wap, nullptr, hiF, 384, 512, 640, 0);
    gemm(fpk, Wbp, nullptr, hjF, 384, 512, 640, 0);

    // pair head
    const float* Wc = tp_w1 + 1280 * 512;
    dim3 pgrid(384, 6);
    pair_kernel<<<pgrid, 256, 0, stream>>>(hiF, hjF, out_dms, out_shape, Wc,
                                           tp_b1, tp_g, tp_be,
                                           w2p, tp_b2, w3p, tp_b3, out_pair);
}
